// Model_31293131718969
// MI455X (gfx1250) — compile-verified, matrix operand order checked by reading, not run on hardware
//
#include <hip/hip_runtime.h>
#include <hip/hip_bf16.h>

// ---------------------------------------------------------------------------
// GCN (2x GCNConv + linear + log_softmax) for MI455X (gfx1250, wave32, WMMA,
// double-buffered Tensor-Data-Mover staging of weight tiles into LDS).
// ---------------------------------------------------------------------------

typedef __attribute__((ext_vector_type(16))) __bf16 v16bf;
typedef __attribute__((ext_vector_type(8)))  float  v8f;
typedef __attribute__((ext_vector_type(4)))  unsigned int u32x4;
typedef __attribute__((ext_vector_type(4)))  int          i32x4;
typedef __attribute__((ext_vector_type(8)))  int          i32x8;

#define N_NODES 100000
#define F_IN    512
#define H1      256
#define H2      128

// ---------------------------------------------------------------------------
// TDM: stage a 2D f32 tile (tileH rows x tileW cols, row stride `stride`
// elements) from global into LDS at byte offset `ldsOff`.
// Descriptor per CDNA5 ISA 8.3/8.4: group0 = {count=1, lds_addr,
// global_addr, type=2}; group1 = {data_size=4B, tensor/tile dims, stride}.
// Issued by one wave (TDM ignores EXEC; TENSORcnt is per-wave).
// ---------------------------------------------------------------------------
#if __has_builtin(__builtin_amdgcn_tensor_load_to_lds)
#define HAVE_TDM 1
__device__ __forceinline__ void tdm_load_tile_f32(const float* gsrc, int tileW,
                                                  int tileH, int stride,
                                                  unsigned ldsOff) {
    const unsigned long long ga = (unsigned long long)gsrc;
    u32x4 g0 = {0u, 0u, 0u, 0u};
    g0.x = 1u;                                           // count=1 (valid user D#)
    g0.y = ldsOff;                                       // lds_addr (bytes)
    g0.z = (unsigned)ga;                                 // global_addr[31:0]
    g0.w = (unsigned)((ga >> 32) & 0x01FFFFFFu) | (2u << 30);  // addr[56:32] | type=2
    i32x8 g1 = {0, 0, 0, 0, 0, 0, 0, 0};
    g1[0] = 2 << 16;                                     // data_size = 4 bytes
    g1[1] = (tileW & 0xFFFF) << 16;                      // tensor_dim0[15:0]
    g1[2] = (tileH & 0xFFFF) << 16;                      // tensor_dim1[15:0]
    g1[3] = (tileW & 0xFFFF) << 16;                      // tile_dim0
    g1[4] = (tileH & 0xFFFF);                            // tile_dim1 (tile_dim2=0)
    g1[5] = stride;                                      // tensor_dim0_stride[31:0]
    i32x4 gz = {0, 0, 0, 0};
#if __clang_major__ >= 23
    i32x8 gz8 = {0, 0, 0, 0, 0, 0, 0, 0};
    __builtin_amdgcn_tensor_load_to_lds(g0, g1, gz, gz, gz8, 0);
#else
    __builtin_amdgcn_tensor_load_to_lds(g0, g1, gz, gz, 0);
#endif
}
__device__ __forceinline__ void tdm_wait() {
#if __has_builtin(__builtin_amdgcn_s_wait_tensorcnt)
    __builtin_amdgcn_s_wait_tensorcnt(0);
#else
    asm volatile("s_wait_tensorcnt 0x0" ::: "memory");
#endif
}
#else
#define HAVE_TDM 0
#endif

// ---------------------------------------------------------------------------
// Generic zero-fill
// ---------------------------------------------------------------------------
__global__ void zero_kernel(float* __restrict__ p, long long n) {
    long long i = (long long)blockIdx.x * blockDim.x + threadIdx.x;
    if (i < n) p[i] = 0.0f;
}

// ---------------------------------------------------------------------------
// Degree count + dinv
// ---------------------------------------------------------------------------
__global__ void deg_count_kernel(const int* __restrict__ dst, float* __restrict__ deg,
                                 long long E) {
    long long i = (long long)blockIdx.x * blockDim.x + threadIdx.x;
    if (i < E) atomicAdd(&deg[dst[i]], 1.0f);
}

__global__ void dinv_kernel(float* __restrict__ deg, int n) {
    int i = blockIdx.x * blockDim.x + threadIdx.x;
    if (i < n) deg[i] = rsqrtf(deg[i] + 1.0f);
}

// ---------------------------------------------------------------------------
// WMMA GEMM, double-buffered TDM weight staging:
//   out[N,H] = A[N,K] (f32->bf16) x W[K,H] (f32->bf16), f32 accumulate.
// Block = 8 waves = 128 rows; blockIdx.y selects a 128-col stripe (HB).
// Pipeline per 64-wide K chunk i:
//   wave0: issue TDM for chunk i+1 -> buf[(i+1)&1]     (overlaps with compute)
//   all  : 2 k-steps x 8 col tiles = 16 v_wmma on buf[i&1] (ds_load B frags)
//   wave0: s_wait_tensorcnt 0 ; all: barrier
// The end-of-iteration barrier also guarantees readers of buf[(i+1)&1]
// (chunk i-1) finished before the next issue overwrites it.
// ---------------------------------------------------------------------------
template <int K, int H, int HB>
__global__ __launch_bounds__(256)
void gemm_bf16_wmma_lds(const float* __restrict__ A, const float* __restrict__ W,
                        float* __restrict__ out, int N) {
    constexpr int KC = 64;                 // K chunk per stage
    constexpr int CT = HB / 16;            // column tiles per wave
    constexpr int NCHUNK = K / KC;
    __shared__ float wtile[2 * KC * HB];   // double buffer, 2 x 32 KB

    const int lane   = threadIdx.x & 31;
    const int wave   = threadIdx.x >> 5;
    const int lrow   = lane & 15;
    const int hi     = lane >> 4;
    const int colOff = blockIdx.y * HB;
    const int rowBase = (blockIdx.x * 8 + wave) * 16;
    const bool active = rowBase < N;       // no early return: barriers below
    const long long arow = (long long)((active ? rowBase : 0) + lrow) * K;

    v8f acc[CT];
#pragma unroll
    for (int t = 0; t < CT; ++t) acc[t] = (v8f){};

#if HAVE_TDM
    // Prologue: stage chunk 0 synchronously.
    if (wave == 0) {
        tdm_load_tile_f32(W + colOff, HB, KC, H, 0u);
        tdm_wait();
    }
    __syncthreads();
#endif

    for (int ch = 0; ch < NCHUNK; ++ch) {
        const int kb = ch * KC;
        const float* buf = wtile + (ch & 1) * (KC * HB);

#if HAVE_TDM
        if (wave == 0 && ch + 1 < NCHUNK) {      // prefetch next chunk via DMA
            tdm_load_tile_f32(W + (long long)(kb + KC) * H + colOff, HB, KC, H,
                              (unsigned)(((ch + 1) & 1) * (KC * HB * 4)));
        }
#else
        __syncthreads();
        {
            const float* wsrc = W + (long long)kb * H + colOff;
            for (int i = threadIdx.x; i < KC * HB / 4; i += 256) {
                const int r = i / (HB / 4), c = i - r * (HB / 4);
                ((float4*)buf)[i] = ((const float4*)(wsrc + (long long)r * H))[c];
            }
        }
        __syncthreads();
#endif

#pragma unroll
        for (int ks = 0; ks < KC; ks += 32) {
            // A fragment: 16 bf16 per lane (16-bit A-matrix K striping)
            v16bf a;
#pragma unroll
            for (int j = 0; j < 8; ++j) {
                const int k = kb + ks + hi * 8 + ((j < 4) ? j * 2 : 16 + (j - 4) * 2);
                a[2 * j]     = (__bf16)A[arow + k];
                a[2 * j + 1] = (__bf16)A[arow + k + 1];
            }
#pragma unroll
            for (int t = 0; t < CT; ++t) {
                v16bf b;
#pragma unroll
                for (int j = 0; j < 8; ++j) {
                    const int k = ks + hi * 8 + ((j < 4) ? j * 2 : 16 + (j - 4) * 2);
                    b[2 * j]     = (__bf16)buf[k * HB + t * 16 + lrow];
                    b[2 * j + 1] = (__bf16)buf[(k + 1) * HB + t * 16 + lrow];
                }
                acc[t] = __builtin_amdgcn_wmma_f32_16x16x32_bf16(
                    false, a, false, b, (short)0, acc[t], false, false);
            }
        }

#if HAVE_TDM
        if (wave == 0) tdm_wait();               // next tile landed
        __syncthreads();                         // all readers of buf done; tile visible
#endif
    }

    if (active) {
#pragma unroll
        for (int t = 0; t < CT; ++t)
#pragma unroll
            for (int r = 0; r < 8; ++r)
                out[(long long)(rowBase + r + 8 * hi) * H + colOff + t * 16 + lrow] =
                    acc[t][r];
    }
}

// ---------------------------------------------------------------------------
// Edge scatter-add: agg[dst] += h[src] * dinv[src]*dinv[dst]
// One thread per (edge, float4-chunk); consecutive threads cover one source
// row -> coalesced 1KB gathers, f32 atomics scatter. This is the HBM-bound
// hot loop (~10 GB total -> ~0.45 ms at 23.3 TB/s).
// ---------------------------------------------------------------------------
__global__ __launch_bounds__(256)
void edge_agg_kernel(const float* __restrict__ h, const int* __restrict__ src,
                     const int* __restrict__ dst, const float* __restrict__ dinv,
                     float* __restrict__ agg, long long E, int H) {
    const long long chunks = (long long)(H >> 2);
    const long long idx = (long long)blockIdx.x * blockDim.x + threadIdx.x;
    if (idx >= E * chunks) return;
    const long long e = idx / chunks;
    const int       c = (int)(idx - e * chunks);

    const int s = src[e];
    const int d = dst[e];
    const float nrm = dinv[s] * dinv[d];

    const float* hp = h + (long long)s * H + c * 4;
    float* ap = agg + (long long)d * H + c * 4;
    __builtin_prefetch(hp, 0, 1);                 // global_prefetch_b8

    const float4 v = *(const float4*)hp;
    atomicAdd(ap + 0, v.x * nrm);
    atomicAdd(ap + 1, v.y * nrm);
    atomicAdd(ap + 2, v.z * nrm);
    atomicAdd(ap + 3, v.w * nrm);
}

// ---------------------------------------------------------------------------
// Self-loop + bias + ReLU: agg = relu(agg + h*dinv^2 + b)
// ---------------------------------------------------------------------------
__global__ void finish_layer_kernel(const float* __restrict__ h,
                                    const float* __restrict__ dinv,
                                    const float* __restrict__ bias,
                                    float* __restrict__ agg, int n, int H) {
    const long long i = (long long)blockIdx.x * blockDim.x + threadIdx.x;
    if (i >= (long long)n * H) return;
    const int node = (int)(i / H);
    const int f    = (int)(i - (long long)node * H);
    const float di = dinv[node];
    float v = agg[i] + h[i] * di * di + bias[f];
    agg[i] = fmaxf(v, 0.0f);
}

// ---------------------------------------------------------------------------
// Head: logits = h[N,128] @ Wl[128,4] + bl, then log_softmax over 4 classes.
// ---------------------------------------------------------------------------
__global__ void head_kernel(const float* __restrict__ h, const float* __restrict__ Wl,
                            const float* __restrict__ bl, float* __restrict__ out, int n) {
    const int node = blockIdx.x * blockDim.x + threadIdx.x;
    if (node >= n) return;
    float a0 = bl[0], a1 = bl[1], a2 = bl[2], a3 = bl[3];
    const float* row = h + (long long)node * H2;
#pragma unroll 8
    for (int k = 0; k < H2; ++k) {
        const float x = row[k];
        const float4 w = *(const float4*)(Wl + k * 4);
        a0 += x * w.x; a1 += x * w.y; a2 += x * w.z; a3 += x * w.w;
    }
    const float m = fmaxf(fmaxf(a0, a1), fmaxf(a2, a3));
    const float e0 = __expf(a0 - m), e1 = __expf(a1 - m);
    const float e2 = __expf(a2 - m), e3 = __expf(a3 - m);
    const float lse = m + __logf(e0 + e1 + e2 + e3);
    float4 o = make_float4(a0 - lse, a1 - lse, a2 - lse, a3 - lse);
    *(float4*)(out + (long long)node * 4) = o;
}

// ---------------------------------------------------------------------------
// Launch. Inputs (setup_inputs order):
//   0:x [N,512] f32  1:edge_index [2,E] int  2:W1 [512,256]  3:b1 [256]
//   4:W2 [256,128]   5:b2 [128]   6:Wl [128,4]   7:bl [4]
// Workspace (floats): dinv[N] | bufA[N*256] (h1, then h2) | bufB[N*256] (agg1/agg2)
// ---------------------------------------------------------------------------
extern "C" void kernel_launch(void* const* d_in, const int* in_sizes, int n_in,
                              void* d_out, int out_size, void* d_ws, size_t ws_size,
                              hipStream_t stream) {
    const float* x   = (const float*)d_in[0];
    const int*   ei  = (const int*)d_in[1];
    const float* W1  = (const float*)d_in[2];
    const float* b1  = (const float*)d_in[3];
    const float* W2  = (const float*)d_in[4];
    const float* b2  = (const float*)d_in[5];
    const float* Wl  = (const float*)d_in[6];
    const float* bl  = (const float*)d_in[7];
    float* out = (float*)d_out;

    const int       n = N_NODES;
    const long long E = (long long)in_sizes[1] / 2;
    const int* src = ei;
    const int* dst = ei + E;

    float* dinv = (float*)d_ws;
    float* bufA = dinv + n;                       // h1 [N,256], later h2 [N,128]
    float* bufB = bufA + (long long)n * H1;       // agg1 [N,256], later agg2 [N,128]

    const int B = 256;
    auto blocks = [](long long work, int b) { return (unsigned)((work + b - 1) / b); };
    const unsigned rowBlocks = blocks((n + 15) / 16, 8);   // 8 waves x 16 rows / block

    // --- degrees -> dinv ---
    zero_kernel<<<blocks(n, B), B, 0, stream>>>(dinv, n);
    deg_count_kernel<<<blocks(E, B), B, 0, stream>>>(dst, dinv, E);
    dinv_kernel<<<blocks(n, B), B, 0, stream>>>(dinv, n);

    // --- layer 1: h1 = x @ W1 ; aggregate ; +self +b1, ReLU ---
    gemm_bf16_wmma_lds<F_IN, H1, 128>
        <<<dim3(rowBlocks, H1 / 128), B, 0, stream>>>(x, W1, bufA, n);
    zero_kernel<<<blocks((long long)n * H1, B), B, 0, stream>>>(bufB, (long long)n * H1);
    edge_agg_kernel<<<blocks(E * (H1 / 4), B), B, 0, stream>>>(bufA, src, dst, dinv,
                                                               bufB, E, H1);
    finish_layer_kernel<<<blocks((long long)n * H1, B), B, 0, stream>>>(bufA, dinv, b1,
                                                                        bufB, n, H1);

    // --- layer 2: h2 = relu(agg1) @ W2 ; aggregate ; +self +b2, ReLU ---
    gemm_bf16_wmma_lds<H1, H2, 128>
        <<<dim3(rowBlocks, H2 / 128), B, 0, stream>>>(bufB, W2, bufA, n);
    zero_kernel<<<blocks((long long)n * H2, B), B, 0, stream>>>(bufB, (long long)n * H2);
    edge_agg_kernel<<<blocks(E * (H2 / 4), B), B, 0, stream>>>(bufA, src, dst, dinv,
                                                               bufB, E, H2);
    finish_layer_kernel<<<blocks((long long)n * H2, B), B, 0, stream>>>(bufA, dinv, b2,
                                                                        bufB, n, H2);

    // --- head: logits + log_softmax ---
    head_kernel<<<blocks(n, B), B, 0, stream>>>(bufB, Wl, bl, out, n);
}
